// Model_6296422056698
// MI455X (gfx1250) — matrix operand order repaired and checked by reading, not compiled, not run on hardware
//
#include <hip/hip_runtime.h>
#include <stdint.h>

// ---------------- WMMA vector types (CDNA5 / gfx1250, wave32) ----------------
typedef __attribute__((ext_vector_type(16))) __bf16 v16bf;
typedef __attribute__((ext_vector_type(8)))  __bf16 v8bf;
typedef __attribute__((ext_vector_type(8)))  float  v8f;

__device__ __forceinline__ unsigned short f2bf(float f) {
    union { float f; unsigned u; } v; v.f = f;
    unsigned u = v.u;
    unsigned r = (u + 0x7FFFu + ((u >> 16) & 1u)) >> 16;  // round-to-nearest-even
    return (unsigned short)r;
}

// ---------------- small utility kernels ----------------
__global__ void zero_kernel(float* p, size_t n) {
    size_t i = (size_t)blockIdx.x * blockDim.x + threadIdx.x;
    if (i < n) p[i] = 0.0f;
}

// norms layout: [r][0]=deg_out(src) [r][1]=deg_in(dst), each Nn floats
__global__ void degree_kernel(const int* __restrict__ src, const int* __restrict__ dst,
                              float* __restrict__ norms, int E, int Nn, int R) {
    int idx = blockIdx.x * blockDim.x + threadIdx.x;
    if (idx >= R * E) return;
    int r = idx / E;
    atomicAdd(&norms[(size_t)(2 * r) * Nn + src[idx]], 1.0f);
    atomicAdd(&norms[(size_t)(2 * r + 1) * Nn + dst[idx]], 1.0f);
}

__global__ void norm_kernel(float* p, size_t n) {
    size_t i = (size_t)blockIdx.x * blockDim.x + threadIdx.x;
    if (i < n) p[i] = rsqrtf(fmaxf(p[i], 1.0f));
}

__global__ void f32_to_bf16_kernel(const float* __restrict__ in, unsigned short* __restrict__ out, size_t n) {
    size_t i = (size_t)blockIdx.x * blockDim.x + threadIdx.x;
    if (i < n) out[i] = f2bf(in[i]);
}

// W: [R][K][Nout] f32  ->  Wt: [R][Nout][K] bf16 (transposed so B-frag loads are contiguous)
__global__ void wtrans_kernel(const float* __restrict__ W, unsigned short* __restrict__ Wt,
                              int K, int Nout, int total) {
    int i = blockIdx.x * blockDim.x + threadIdx.x;
    if (i >= total) return;
    int per = K * Nout;
    int r = i / per;
    int rem = i - r * per;
    int k = rem / Nout;
    int n = rem - k * Nout;
    Wt[(size_t)(r * Nout + n) * K + k] = f2bf(W[i]);
}

// ---------------- WMMA GEMM: C[M,Nout] = A[M,K](bf16) * Bt[Nout,K]^T(bf16) ----------------
// One wave per 16x16 C tile; K stepped by 32 via v_wmma_f32_16x16x32_bf16.
// Fragment layouts per CDNA5 ISA 7.12.2:
//   A 16x32: lane(l): m=l&15, h=l>>4; vec[i]=A[m][h*8+i], vec[8+i]=A[m][16+h*8+i]
//   B 32x16: lane(l): n=l&15, h=l>>4; vec[i]=Bt[n][h*8+i], vec[8+i]=Bt[n][16+h*8+i]
//   C 16x16: lane(l): n=l&15, h=l>>4; acc[v]=C[h*8+v][n]
__global__ void gemm_bf16_wmma(const unsigned short* __restrict__ A,
                               const unsigned short* __restrict__ Bt,
                               float* __restrict__ C, int Nout, int K) {
    int lane = threadIdx.x;            // 0..31 (wave32)
    int m0 = blockIdx.x * 16;
    int n0 = blockIdx.y * 16;
    int mm = lane & 15;
    int hh = lane >> 4;

    const __bf16* Arow = (const __bf16*)(A + (size_t)(m0 + mm) * K);
    const __bf16* Brow = (const __bf16*)(Bt + (size_t)(n0 + mm) * K);

    v8f acc = {};
    for (int k0 = 0; k0 < K; k0 += 32) {
        v8bf alo = *(const v8bf*)(Arow + k0 + hh * 8);
        v8bf ahi = *(const v8bf*)(Arow + k0 + 16 + hh * 8);
        v16bf a = __builtin_shufflevector(alo, ahi, 0,1,2,3,4,5,6,7,8,9,10,11,12,13,14,15);
        v8bf blo = *(const v8bf*)(Brow + k0 + hh * 8);
        v8bf bhi = *(const v8bf*)(Brow + k0 + 16 + hh * 8);
        v16bf b = __builtin_shufflevector(blo, bhi, 0,1,2,3,4,5,6,7,8,9,10,11,12,13,14,15);
        acc = __builtin_amdgcn_wmma_f32_16x16x32_bf16(false, a, false, b,
                                                      (short)0, acc, false, false);
    }
    float* Cp = C + (size_t)(m0 + hh * 8) * Nout + (n0 + mm);
#pragma unroll
    for (int v = 0; v < 8; ++v) Cp[(size_t)v * Nout] = acc[v];
}

// ---------------- edge scatter with fused src/dst norms ----------------
// agg[dst] += xw[src] * norm_src[src] * norm_dst[dst]   (one block per edge, one thread per feature)
__global__ void scatter_kernel(const float* __restrict__ xw, const int* __restrict__ src,
                               const int* __restrict__ dst, const float* __restrict__ nsrc,
                               const float* __restrict__ ndst, float* __restrict__ agg, int F) {
    int e = blockIdx.x;
    int f = threadIdx.x;
    int s = src[e], d = dst[e];
    float c = nsrc[s] * ndst[d];
    atomicAdd(&agg[(size_t)d * F + f], xw[(size_t)s * F + f] * c);
}

// h = relu(agg + sum_r b[r]) -> bf16 (feeds layer-2 WMMA)
__global__ void relu_bias_bf16_kernel(const float* __restrict__ agg, const float* __restrict__ b,
                                      unsigned short* __restrict__ out, int F, int R, size_t n) {
    size_t i = (size_t)blockIdx.x * blockDim.x + threadIdx.x;
    if (i >= n) return;
    int f = (int)(i % F);
    float bs = 0.0f;
    for (int r = 0; r < R; ++r) bs += b[(size_t)r * F + f];
    out[i] = f2bf(fmaxf(agg[i] + bs, 0.0f));
}

// h2 = agg + sum_r b[r]  (fp32, final node embeddings)
__global__ void bias_kernel(const float* __restrict__ agg, const float* __restrict__ b,
                            float* __restrict__ out, int F, int R, size_t n) {
    size_t i = (size_t)blockIdx.x * blockDim.x + threadIdx.x;
    if (i >= n) return;
    int f = (int)(i % F);
    float bs = 0.0f;
    for (int r = 0; r < R; ++r) bs += b[(size_t)r * F + f];
    out[i] = agg[i] + bs;
}

// one wave per edge: out[e] = dot(h[si[e]], h[di[e]])
__global__ void dot_kernel(const float* __restrict__ h, const int* __restrict__ si,
                           const int* __restrict__ di, float* __restrict__ out, int E, int F) {
    int gid = blockIdx.x * blockDim.x + threadIdx.x;
    int e = gid >> 5;
    int lane = gid & 31;
    if (e >= E) return;
    const float* a = h + (size_t)si[e] * F;
    const float* b = h + (size_t)di[e] * F;
    float acc = 0.0f;
    for (int f = lane; f < F; f += 32) acc += a[f] * b[f];
    for (int off = 16; off > 0; off >>= 1) acc += __shfl_down(acc, off, 32);
    if (lane == 0) out[e] = acc;
}

// ---------------- host orchestration ----------------
extern "C" void kernel_launch(void* const* d_in, const int* in_sizes, int n_in,
                              void* d_out, int out_size, void* d_ws, size_t ws_size,
                              hipStream_t stream) {
    const float* x  = (const float*)d_in[0];
    const int* src  = (const int*)d_in[1];   // [R][E]
    const int* dst  = (const int*)d_in[2];   // [R][E]
    const int* nsi  = (const int*)d_in[3];   // neg_src [E]
    const int* ndi  = (const int*)d_in[4];   // neg_dst [E]
    const float* W1 = (const float*)d_in[5]; // [R][din][dhid]
    const float* b1 = (const float*)d_in[6]; // [R][dhid]
    const float* W2 = (const float*)d_in[7]; // [R][dhid][dout]
    const float* b2 = (const float*)d_in[8]; // [R][dout]

    const int E    = in_sizes[3];
    const int R    = in_sizes[1] / E;
    const int dhid = in_sizes[6] / R;
    const int dout = in_sizes[8] / R;
    const int din  = in_sizes[5] / (R * dhid);
    const int Nn   = in_sizes[0] / din;      // 50000 (multiple of 16)

    // workspace carve-out (floats)
    float* w = (float*)d_ws;
    size_t off = 0;
    float* norms = w + off; off += (size_t)2 * R * Nn;
    float* agg1  = w + off; off += (size_t)Nn * dhid;
    float* agg2  = w + off; off += (size_t)Nn * dout;
    size_t zeroCount = off;                               // zero norms+agg1+agg2 each call
    float* xw    = w + off; off += (size_t)Nn * dhid;     // reused for layer-2 (dout<=dhid)
    float* h2    = w + off; off += (size_t)Nn * dout;
    unsigned short* xbf = (unsigned short*)(w + off); off += ((size_t)Nn * din + 1) / 2;
    unsigned short* hbf = (unsigned short*)(w + off); off += ((size_t)Nn * dhid + 1) / 2;
    unsigned short* w1t = (unsigned short*)(w + off); off += ((size_t)R * din * dhid + 1) / 2;
    unsigned short* w2t = (unsigned short*)(w + off); off += ((size_t)R * dhid * dout + 1) / 2;

    const int T = 256;
    zero_kernel<<<(unsigned)((zeroCount + T - 1) / T), T, 0, stream>>>(w, zeroCount);
    degree_kernel<<<(unsigned)(((size_t)R * E + T - 1) / T), T, 0, stream>>>(src, dst, norms, E, Nn, R);
    norm_kernel<<<(unsigned)(((size_t)2 * R * Nn + T - 1) / T), T, 0, stream>>>(norms, (size_t)2 * R * Nn);
    f32_to_bf16_kernel<<<(unsigned)(((size_t)Nn * din + T - 1) / T), T, 0, stream>>>(x, xbf, (size_t)Nn * din);
    wtrans_kernel<<<(unsigned)(((size_t)R * din * dhid + T - 1) / T), T, 0, stream>>>(W1, w1t, din, dhid, R * din * dhid);
    wtrans_kernel<<<(unsigned)(((size_t)R * dhid * dout + T - 1) / T), T, 0, stream>>>(W2, w2t, dhid, dout, R * dhid * dout);

    // ---- layer 1 ----
    dim3 g1((unsigned)(Nn / 16), (unsigned)(dhid / 16));
    for (int r = 0; r < R; ++r) {
        gemm_bf16_wmma<<<g1, 32, 0, stream>>>(xbf, w1t + (size_t)r * din * dhid, xw, dhid, din);
        scatter_kernel<<<(unsigned)E, dhid, 0, stream>>>(xw, src + (size_t)r * E, dst + (size_t)r * E,
                                                         norms + (size_t)(2 * r) * Nn,
                                                         norms + (size_t)(2 * r + 1) * Nn, agg1, dhid);
    }
    relu_bias_bf16_kernel<<<(unsigned)(((size_t)Nn * dhid + T - 1) / T), T, 0, stream>>>(
        agg1, b1, hbf, dhid, R, (size_t)Nn * dhid);

    // ---- layer 2 ----
    dim3 g2((unsigned)(Nn / 16), (unsigned)(dout / 16));
    for (int r = 0; r < R; ++r) {
        gemm_bf16_wmma<<<g2, 32, 0, stream>>>(hbf, w2t + (size_t)r * dhid * dout, xw, dout, dhid);
        scatter_kernel<<<(unsigned)E, dout, 0, stream>>>(xw, src + (size_t)r * E, dst + (size_t)r * E,
                                                         norms + (size_t)(2 * r) * Nn,
                                                         norms + (size_t)(2 * r + 1) * Nn, agg2, dout);
    }
    bias_kernel<<<(unsigned)(((size_t)Nn * dout + T - 1) / T), T, 0, stream>>>(
        agg2, b2, h2, dout, R, (size_t)Nn * dout);

    // ---- edge scoring: pos then neg ----
    float* out = (float*)d_out;
    size_t lanesTot = (size_t)E * 32;
    dot_kernel<<<(unsigned)((lanesTot + T - 1) / T), T, 0, stream>>>(h2, src, dst, out, E, dout);
    dot_kernel<<<(unsigned)((lanesTot + T - 1) / T), T, 0, stream>>>(h2, nsi, ndi, out + E, E, dout);
}
